// GATEncoder_19851338842496
// MI455X (gfx1250) — matrix operand order repaired and checked by reading, not compiled, not run on hardware
//
#include <hip/hip_runtime.h>
#include <math.h>

typedef __attribute__((ext_vector_type(16))) _Float16 v16h;
typedef __attribute__((ext_vector_type(8)))  float    v8f;

#define HEADS 4
#define HID   32
#define FDIM  128   // IN_DIM == HEADS*HID == 128
#define NEG_SLOPE 0.2f

// ---------------------------------------------------------------------------
// GEMM: h = x @ W   (x:[N,128] f32, W:[128,128] f32 row-major, h:[N,128] f32)
// One wave computes a 16x128 output strip via 8 n-tiles x 4 k-chunks of
// v_wmma_f32_16x16x32_f16.
// W is staged to LDS as f16 *transposed* (sWt[n][k]) so each lane's
// B-fragment (16 consecutive k at fixed n) is one contiguous 32-byte,
// 32B-aligned LDS read -> two ds_load_b128 per fragment.
// ---------------------------------------------------------------------------
__global__ void __launch_bounds__(128)
gat_gemm_kernel(const float* __restrict__ x, const float* __restrict__ W,
                float* __restrict__ h, int nrows16) {
  __shared__ __align__(32) _Float16 sWt[FDIM * FDIM];  // 32 KB of 320 KB LDS
  const int tid = threadIdx.x;
  // stage transposed: sWt[n*128 + k] = W[k*128 + n]; global reads coalesced
  for (int i = tid; i < FDIM * FDIM; i += 128) {
    const int k = i >> 7, n = i & (FDIM - 1);
    sWt[n * FDIM + k] = (_Float16)W[i];
  }
  __syncthreads();

  const int wave = tid >> 5;
  const int lane = tid & 31;
  const int lo = lane & 15, hi = lane >> 4;
  const int tile = blockIdx.x * 4 + wave;
  if (tile >= nrows16) return;          // wave-uniform: EXEC stays all-ones
  const int m0 = tile * 16;

  // A fragments: 16-bit A 16x32 layout. Lane(lo,hi): row M=lo,
  // element j -> K = hi*16 + j (matches B's contiguous k order).
  v16h afr[4];
  const float* arow = x + (size_t)(m0 + lo) * FDIM;
#pragma unroll
  for (int kc = 0; kc < 4; ++kc) {
    const float* ap = arow + kc * 32;
#pragma unroll
    for (int j = 0; j < 8; ++j) {
      afr[kc][j]     = (_Float16)ap[hi * 16 + j];
      afr[kc][8 + j] = (_Float16)ap[hi * 16 + 8 + j];
    }
  }

  v8f acc[8];
#pragma unroll
  for (int nt = 0; nt < 8; ++nt)
    acc[nt] = (v8f){0.f, 0.f, 0.f, 0.f, 0.f, 0.f, 0.f, 0.f};

#pragma unroll
  for (int nt = 0; nt < 8; ++nt) {
    const int n0 = nt * 16;
    // per-lane base: column n0+lo of W, i.e. 128 consecutive k values
    const _Float16* bcol = sWt + (size_t)(n0 + lo) * FDIM;
#pragma unroll
    for (int kc = 0; kc < 4; ++kc) {
      // B 32x16 layout: lane(lo,hi): col N=lo+n0, element j -> K = kc*32+hi*16+j
      // 16 consecutive f16 = 32 contiguous bytes, 32B aligned.
      const v16h b = *(const v16h*)(bcol + kc * 32 + hi * 16);
      acc[nt] = __builtin_amdgcn_wmma_f32_16x16x32_f16(
          false, afr[kc], false, b, (short)0, acc[nt], false, false);
    }
  }

  // D layout: VGPR i -> row m0 + hi*8 + i, col n0 + lo
#pragma unroll
  for (int nt = 0; nt < 8; ++nt) {
#pragma unroll
    for (int i = 0; i < 8; ++i) {
      h[(size_t)(m0 + hi * 8 + i) * FDIM + nt * 16 + lo] = acc[nt][i];
    }
  }
}

// ---------------------------------------------------------------------------
// Attention coefficients: a_src[n,h] = <h[n,h,:], att_src[h,:]>, same for dst
// ---------------------------------------------------------------------------
__global__ void gat_attn_kernel(const float* __restrict__ h,
                                const float* __restrict__ att_src,
                                const float* __restrict__ att_dst,
                                float* __restrict__ a_src,
                                float* __restrict__ a_dst, int n_nodes) {
  int t = blockIdx.x * blockDim.x + threadIdx.x;
  if (t >= n_nodes * HEADS) return;
  int n = t >> 2, hh = t & 3;
  const float* hp = h + (size_t)n * FDIM + hh * HID;
  const float* as = att_src + hh * HID;
  const float* ad = att_dst + hh * HID;
  float s0 = 0.f, s1 = 0.f;
#pragma unroll
  for (int c = 0; c < HID; ++c) {
    float v = hp[c];
    s0 += v * as[c];
    s1 += v * ad[c];
  }
  a_src[t] = s0;
  a_dst[t] = s1;
}

// ---------------------------------------------------------------------------
// Per-layer init: acc = 0, e_max = -inf, denom = 0
// ---------------------------------------------------------------------------
__global__ void gat_init_kernel(float* __restrict__ acc,
                                float* __restrict__ e_max,
                                float* __restrict__ denom, int n_nodes) {
  int t = blockIdx.x * blockDim.x + threadIdx.x;
  if (t < n_nodes * FDIM) acc[t] = 0.f;
  if (t < n_nodes * HEADS) {
    e_max[t] = -__builtin_inff();
    denom[t] = 0.f;
  }
}

// float atomic max via signed/unsigned int ordering trick (race-safe)
__device__ __forceinline__ void atomicMaxF32(float* addr, float v) {
  if (v >= 0.f)
    atomicMax((int*)addr, __float_as_int(v));
  else
    atomicMin((unsigned int*)addr, (unsigned int)__float_as_int(v));
}

// ---------------------------------------------------------------------------
// Edge pass 1: e = leaky_relu(a_src[src] + a_dst[dst]); segment max over dst
// ---------------------------------------------------------------------------
__global__ void gat_edge_max_kernel(const int* __restrict__ src,
                                    const int* __restrict__ dst,
                                    const float* __restrict__ a_src,
                                    const float* __restrict__ a_dst,
                                    float* __restrict__ e_buf,
                                    float* __restrict__ e_max, int n_edges) {
  int e = blockIdx.x * blockDim.x + threadIdx.x;
  if (e >= n_edges) return;
  int s = src[e], d = dst[e];
#pragma unroll
  for (int hh = 0; hh < HEADS; ++hh) {
    float v = a_src[(size_t)s * HEADS + hh] + a_dst[(size_t)d * HEADS + hh];
    v = v > 0.f ? v : NEG_SLOPE * v;
    e_buf[(size_t)e * HEADS + hh] = v;
    atomicMaxF32(e_max + (size_t)d * HEADS + hh, v);
  }
}

// ---------------------------------------------------------------------------
// Edge pass 2: ex = exp(e - max[dst]); segment sum -> denom
// ---------------------------------------------------------------------------
__global__ void gat_edge_exp_kernel(const int* __restrict__ dst,
                                    const float* __restrict__ e_max,
                                    float* __restrict__ e_buf,
                                    float* __restrict__ denom, int n_edges) {
  int e = blockIdx.x * blockDim.x + threadIdx.x;
  if (e >= n_edges) return;
  int d = dst[e];
#pragma unroll
  for (int hh = 0; hh < HEADS; ++hh) {
    float ex = __expf(e_buf[(size_t)e * HEADS + hh] -
                      e_max[(size_t)d * HEADS + hh]);
    e_buf[(size_t)e * HEADS + hh] = ex;
    atomicAdd(denom + (size_t)d * HEADS + hh, ex);
  }
}

// ---------------------------------------------------------------------------
// Scatter: one wave per edge; lane handles 4 channels (float4 gather),
// head = lane/8; acc[dst] += alpha * h[src]
// ---------------------------------------------------------------------------
__global__ void gat_scatter_kernel(const int* __restrict__ src,
                                   const int* __restrict__ dst,
                                   const float* __restrict__ h,
                                   const float* __restrict__ e_buf,
                                   const float* __restrict__ denom,
                                   float* __restrict__ acc, int n_edges) {
  long g = (long)blockIdx.x * blockDim.x + threadIdx.x;
  int e = (int)(g >> 5);
  int lane = (int)(g & 31);
  if (e >= n_edges) return;
  int s = src[e], d = dst[e];
  int hh = lane >> 3;
  float alpha = e_buf[(size_t)e * HEADS + hh] /
                denom[(size_t)d * HEADS + hh];
  const float4 hv = *(const float4*)(h + (size_t)s * FDIM + lane * 4);
  float* ap = acc + (size_t)d * FDIM + lane * 4;
  atomicAdd(ap + 0, hv.x * alpha);
  atomicAdd(ap + 1, hv.y * alpha);
  atomicAdd(ap + 2, hv.z * alpha);
  atomicAdd(ap + 3, hv.w * alpha);
}

// ---------------------------------------------------------------------------
// Finalize: v = relu(acc + bias); write to strided output slot; optionally
// keep activations in acc (layer 1 feeds layer 2's GEMM).
// ---------------------------------------------------------------------------
__global__ void gat_finalize_kernel(float* __restrict__ acc,
                                    const float* __restrict__ bias,
                                    float* __restrict__ out, int slot,
                                    int n_nodes, int keep) {
  int t = blockIdx.x * blockDim.x + threadIdx.x;
  if (t >= n_nodes * FDIM) return;
  int f = t & (FDIM - 1);
  float v = acc[t] + bias[f];
  v = v > 0.f ? v : 0.f;
  if (keep) acc[t] = v;
  out[(size_t)(t >> 7) * (2 * FDIM) + f * 2 + slot] = v;
}

// ---------------------------------------------------------------------------
extern "C" void kernel_launch(void* const* d_in, const int* in_sizes, int n_in,
                              void* d_out, int out_size, void* d_ws,
                              size_t ws_size, hipStream_t stream) {
  (void)n_in; (void)out_size; (void)ws_size;
  const float* x   = (const float*)d_in[0];
  const int*   ei  = (const int*)d_in[1];
  const float* W1  = (const float*)d_in[2];
  const float* as1 = (const float*)d_in[3];
  const float* ad1 = (const float*)d_in[4];
  const float* b1  = (const float*)d_in[5];
  const float* W2  = (const float*)d_in[6];
  const float* as2 = (const float*)d_in[7];
  const float* ad2 = (const float*)d_in[8];
  const float* b2  = (const float*)d_in[9];

  const int n_nodes = in_sizes[0] / FDIM;
  const int n_edges = in_sizes[1] / 2;
  const int* srcp = ei;
  const int* dstp = ei + n_edges;
  float* out = (float*)d_out;

  // workspace layout (~134 MB total)
  float* ws      = (float*)d_ws;
  float* buf_h   = ws; ws += (size_t)n_nodes * FDIM;   // 51.2 MB
  float* buf_acc = ws; ws += (size_t)n_nodes * FDIM;   // 51.2 MB
  float* a_src   = ws; ws += (size_t)n_nodes * HEADS;
  float* a_dst   = ws; ws += (size_t)n_nodes * HEADS;
  float* e_max   = ws; ws += (size_t)n_nodes * HEADS;
  float* denom   = ws; ws += (size_t)n_nodes * HEADS;
  float* e_buf   = ws; ws += (size_t)n_edges * HEADS;  // 25.6 MB

  const dim3 blk(256);
  const int nt16        = n_nodes / 16;                 // 100000/16 = 6250
  const int gemm_blocks = (nt16 + 3) / 4;
  const int nf = n_nodes * FDIM;
  const int nh = n_nodes * HEADS;
  const long scat_thr   = (long)n_edges * 32;
  const int  scat_blk   = (int)((scat_thr + 255) / 256);

  // ---------------- layer 1 ----------------
  gat_gemm_kernel<<<gemm_blocks, 128, 0, stream>>>(x, W1, buf_h, nt16);
  gat_attn_kernel<<<(nh + 255) / 256, blk, 0, stream>>>(buf_h, as1, ad1,
                                                        a_src, a_dst, n_nodes);
  gat_init_kernel<<<(nf + 255) / 256, blk, 0, stream>>>(buf_acc, e_max, denom,
                                                        n_nodes);
  gat_edge_max_kernel<<<(n_edges + 255) / 256, blk, 0, stream>>>(
      srcp, dstp, a_src, a_dst, e_buf, e_max, n_edges);
  gat_edge_exp_kernel<<<(n_edges + 255) / 256, blk, 0, stream>>>(
      dstp, e_max, e_buf, denom, n_edges);
  gat_scatter_kernel<<<scat_blk, blk, 0, stream>>>(srcp, dstp, buf_h, e_buf,
                                                   denom, buf_acc, n_edges);
  gat_finalize_kernel<<<(nf + 255) / 256, blk, 0, stream>>>(
      buf_acc, b1, out, /*slot=*/0, n_nodes, /*keep=*/1);

  // ---------------- layer 2 (input = buf_acc = x1) ----------------
  gat_gemm_kernel<<<gemm_blocks, 128, 0, stream>>>(buf_acc, W2, buf_h, nt16);
  gat_attn_kernel<<<(nh + 255) / 256, blk, 0, stream>>>(buf_h, as2, ad2,
                                                        a_src, a_dst, n_nodes);
  // buf_acc (x1) already consumed by gemm above; safe to recycle
  gat_init_kernel<<<(nf + 255) / 256, blk, 0, stream>>>(buf_acc, e_max, denom,
                                                        n_nodes);
  gat_edge_max_kernel<<<(n_edges + 255) / 256, blk, 0, stream>>>(
      srcp, dstp, a_src, a_dst, e_buf, e_max, n_edges);
  gat_edge_exp_kernel<<<(n_edges + 255) / 256, blk, 0, stream>>>(
      dstp, e_max, e_buf, denom, n_edges);
  gat_scatter_kernel<<<scat_blk, blk, 0, stream>>>(srcp, dstp, buf_h, e_buf,
                                                   denom, buf_acc, n_edges);
  gat_finalize_kernel<<<(nf + 255) / 256, blk, 0, stream>>>(
      buf_acc, b2, out, /*slot=*/1, n_nodes, /*keep=*/0);
}
